// Tranformer_49692771614971
// MI455X (gfx1250) — matrix operand order repaired and checked by reading, not compiled, not run on hardware
//
#include <hip/hip_runtime.h>

#define DEV __device__ __forceinline__

typedef __attribute__((ext_vector_type(16))) __bf16 v16bf;
typedef __attribute__((ext_vector_type(8)))  __bf16 v8bf;
typedef __attribute__((ext_vector_type(8)))  float  v8f;
typedef __attribute__((ext_vector_type(4)))  unsigned int u32x4;
typedef __attribute__((ext_vector_type(8)))  int i32x8;
typedef __attribute__((ext_vector_type(4)))  int i32x4;

constexpr int L_ = 6, B_ = 2, S_ = 2048, D_ = 512, H_ = 8, DH_ = 64, FF_ = 2048;
constexpr int BS_ = B_ * S_;

#if __has_builtin(__builtin_amdgcn_tensor_load_to_lds)
#define HAVE_TDM 1
#endif

DEV unsigned short f2bf(float f) {
  unsigned int u = __builtin_bit_cast(unsigned int, f);
  u += 0x7FFFu + ((u >> 16) & 1u);   // round-to-nearest-even
  return (unsigned short)(u >> 16);
}

DEV v8f wmma_bf16(v16bf a, v16bf b, v8f c) {
  return __builtin_amdgcn_wmma_f32_16x16x32_bf16(false, a, false, b, (short)0, c,
                                                 false, false);
}

// --------------------------------------------------------------- TDM staging
// Stage a [32 x TK] bf16 tile (row stride K elements) from global into LDS.
// Issued by one wave; completion = s_wait_tensorcnt 0 + workgroup barrier.
DEV void tdm_stage(unsigned short* ldst, const __bf16* gsrc, int K, int TK,
                   int lane) {
#ifdef HAVE_TDM
  unsigned long long ga = (unsigned long long)(const void*)gsrc;
  u32x4 g0 = {};
  g0[0] = 1u;                                          // count=1, user mode
  g0[1] = (unsigned)(unsigned long long)ldst;          // LDS offset (addr[31:0])
  g0[2] = (unsigned)ga;                                // global_addr[31:0]
  g0[3] = (unsigned)((ga >> 32) & 0x01FFFFFFu)         // global_addr[56:32]
          | 0x80000000u;                               // type=2 ("image")
  i32x8 g1 = {};
  g1[0] = 1 << 16;                                     // data_size=1 -> 2 bytes
  g1[1] = (K & 0xFFFF) << 16;                          // tensor_dim0 lo
  g1[2] = (int)(((unsigned)K >> 16) | (32u << 16));    // dim0 hi | tensor_dim1=32
  g1[3] = TK << 16;                                    // tile_dim0 = TK
  g1[4] = 32;                                          // tile_dim1 = 32 rows
  g1[5] = K;                                           // tensor_dim0_stride lo
  i32x4 z4 = {};
  i32x8 z8 = {};
  (void)lane;
  // 6-arg form (upstream clang-23 / therock-10.0): groups 0..3 + extra + cpol
  __builtin_amdgcn_tensor_load_to_lds(g0, g1, z4, z4, z8, 0);
#else
  // fallback: cooperative copy by the issuing wave (16B per lane-iteration)
  const int chunks = 32 * TK / 8;
  for (int i = lane; i < chunks; i += 32) {
    int r = i / (TK / 8), cc = (i % (TK / 8)) * 8;
    *(v8bf*)(ldst + r * TK + cc) = *(const v8bf*)(gsrc + (size_t)r * K + cc);
  }
#endif
}

DEV void tdm_fence() {
#ifdef HAVE_TDM
  __builtin_amdgcn_s_wait_tensorcnt(0);
#endif
}

// ---------------------------------------------------------------- fp32 -> bf16
__global__ void cvt_bf16_kernel(const float* __restrict__ in,
                                unsigned short* __restrict__ out, long long n) {
  long long i = (long long)blockIdx.x * blockDim.x + threadIdx.x;
  if (i < n) out[i] = f2bf(in[i]);
}

// ---------------------------------------------------------------- WMMA GEMM
// C[M,N] = A[M,K] * W[N,K]^T + bias[N].
// Block = 8 waves, block tile = 32 (M) x 256 (N); wave tile = 32x32 (4 accs).
// A strip (32 rows) is staged to LDS by the TDM (double buffered, TK=128);
// B fragments stream from global (weights live in the 192MB L2).
// Fragment layouts used here (contiguous k order, same for A and B):
//   A frag: lane L -> row L&15, K range (L>>4)*16..+15
//   B frag: lane L -> col L&15, K range (L>>4)*16..+15
//   C tile: lane L, vgpr r -> (M = r + 8*(L>>4), N = L&15)
// MODE: 1=Q head (scaled), 2=K head, 3=V head transposed, 4=raw f32, 5=leaky bf16
template <int MODE>
__global__ __launch_bounds__(256) void gemm_wmma(
    const __bf16* __restrict__ A, const __bf16* __restrict__ W,
    const float* __restrict__ bias, float* __restrict__ outF,
    unsigned short* __restrict__ outB, int M, int N, int K, float scale) {
  constexpr int TK = 128;
  __shared__ unsigned short lds_a[2][32 * TK];   // 2 x 8KB double buffer
  const int lane = threadIdx.x & 31;
  const int wave = threadIdx.x >> 5;
  const int half = lane >> 4;
  const int l15  = lane & 15;
  const int n0 = (blockIdx.x * 8 + wave) * 32;
  const int m0 = blockIdx.y * 32;

  const __bf16* Arow = A + (size_t)m0 * K;
  const __bf16* Wp0 = W + (size_t)(n0 + l15) * K + half * 16;
  const __bf16* Wp1 = W + (size_t)(n0 + 16 + l15) * K + half * 16;

  v8f acc[4] = {};   // [mi*2 + ni]
  const int nch = K / TK;

  if (wave == 0) {
    tdm_stage(&lds_a[0][0], Arow, K, TK, lane);
    tdm_fence();
  }
  __syncthreads();

  for (int c = 0; c < nch; ++c) {
    const int cur = c & 1;
    if (wave == 0 && (c + 1) < nch)        // prefetch next chunk via TDM
      tdm_stage(&lds_a[cur ^ 1][0], Arow + (c + 1) * TK, K, TK, lane);

    const unsigned short* lbase = &lds_a[cur][0];
#pragma unroll
    for (int ks = 0; ks < TK / 32; ++ks) {
      const int kcol = ks * 32 + half * 16;
      v16bf a0, a1;
      ((v8bf*)&a0)[0] = *(const v8bf*)(lbase + (size_t)l15 * TK + kcol);
      ((v8bf*)&a0)[1] = *(const v8bf*)(lbase + (size_t)l15 * TK + kcol + 8);
      ((v8bf*)&a1)[0] = *(const v8bf*)(lbase + (size_t)(16 + l15) * TK + kcol);
      ((v8bf*)&a1)[1] = *(const v8bf*)(lbase + (size_t)(16 + l15) * TK + kcol + 8);
      const int kg = c * TK + ks * 32;
      v16bf b0 = *(const v16bf*)(Wp0 + kg);
      v16bf b1 = *(const v16bf*)(Wp1 + kg);
      __builtin_prefetch(Wp0 + kg + 64, 0, 1);
      __builtin_prefetch(Wp1 + kg + 64, 0, 1);
      acc[0] = wmma_bf16(a0, b0, acc[0]);
      acc[1] = wmma_bf16(a0, b1, acc[1]);
      acc[2] = wmma_bf16(a1, b0, acc[2]);
      acc[3] = wmma_bf16(a1, b1, acc[3]);
    }
    if (wave == 0) tdm_fence();            // TDM done before buffer reuse
    __syncthreads();
  }

  const float bv[2] = {bias[n0 + l15], bias[n0 + 16 + l15]};
#pragma unroll
  for (int mi = 0; mi < 2; ++mi)
#pragma unroll
    for (int ni = 0; ni < 2; ++ni) {
      const v8f a = acc[mi * 2 + ni];
      const int col = n0 + ni * 16 + l15;
#pragma unroll
      for (int r = 0; r < 8; ++r) {
        int row = m0 + mi * 16 + r + 8 * half;
        float v = a[r] + bv[ni];
        if constexpr (MODE == 1 || MODE == 2) {      // [B,H,S,DH]
          if constexpr (MODE == 1) v *= scale;       // q * DH^-0.5
          int bb = row >> 11, ss = row & (S_ - 1);
          int hh = col >> 6,  dd = col & (DH_ - 1);
          outB[(((size_t)bb * H_ + hh) * S_ + ss) * DH_ + dd] = f2bf(v);
        } else if constexpr (MODE == 3) {            // V transposed [B,H,DH,S]
          int bb = row >> 11, ss = row & (S_ - 1);
          int hh = col >> 6,  dd = col & (DH_ - 1);
          outB[(((size_t)bb * H_ + hh) * DH_ + dd) * S_ + ss] = f2bf(v);
        } else if constexpr (MODE == 4) {            // raw f32 for LN epilogue
          outF[(size_t)row * N + col] = v;
        } else {                                     // leaky relu, bf16
          v = v > 0.f ? v : 0.01f * v;
          outB[(size_t)row * N + col] = f2bf(v);
        }
      }
    }
}

// ---------------------------------------------------------------- attention
// One wave per (b, h, 16-query tile). Flash-style online softmax, 32 keys/iter.
// Q: [B,H,S,DH] bf16 (pre-scaled), K: [B,H,S,DH] bf16, Vt: [B,H,DH,S] bf16.
__global__ __launch_bounds__(256) void attn_wmma(
    const __bf16* __restrict__ Q, const __bf16* __restrict__ Kc,
    const __bf16* __restrict__ Vt, const int* __restrict__ kmask,
    unsigned short* __restrict__ O) {
  __shared__ unsigned short lds_p[8][16 * 32];   // 1KB per wave (P staging)
  const int lane = threadIdx.x & 31;
  const int wave = threadIdx.x >> 5;
  const int half = lane >> 4;
  const int l15  = lane & 15;
  const int w  = blockIdx.x * 8 + wave;
  const int QT = S_ / 16;
  const int qt = w % QT;
  const int bh = w / QT;
  const int h  = bh % H_;
  const int b  = bh / H_;

  const int qrow = qt * 16 + l15;
  const __bf16* qbase = Q + (((size_t)b * H_ + h) * S_ + qrow) * DH_;
  v16bf aq[2];
#pragma unroll
  for (int c = 0; c < 2; ++c) {
    v8bf* hh_ = (v8bf*)&aq[c];
    hh_[0] = *(const v8bf*)(qbase + c * 32 + half * 16);
    hh_[1] = *(const v8bf*)(qbase + c * 32 + half * 16 + 8);
  }
  const __bf16* kbase = Kc + ((size_t)b * H_ + h) * S_ * DH_;
  const __bf16* vbase = Vt + ((size_t)b * H_ + h) * DH_ * S_;
  const int* km = kmask + b * S_;

  v8f oacc[4] = {};
  float mrun[8], lrun[8];
#pragma unroll
  for (int r = 0; r < 8; ++r) { mrun[r] = -1e30f; lrun[r] = 0.f; }

  for (int j = 0; j < S_; j += 32) {
    v8f sv[2];
#pragma unroll
    for (int t = 0; t < 2; ++t) {
      int kn = j + t * 16 + l15;
      const __bf16* kp = kbase + (size_t)kn * DH_ + half * 16;
      v16bf b0 = *(const v16bf*)(kp);
      v16bf b1 = *(const v16bf*)(kp + 32);
      v8f st = {};
      st = wmma_bf16(aq[0], b0, st);
      st = wmma_bf16(aq[1], b1, st);
      if (km[kn] == 0) {
#pragma unroll
        for (int r = 0; r < 8; ++r) st[r] = -1e30f;
      }
      sv[t] = st;
    }
    float fac[8];
#pragma unroll
    for (int r = 0; r < 8; ++r) {
      float mr = fmaxf(sv[0][r], sv[1][r]);
#pragma unroll
      for (int x = 1; x < 16; x <<= 1) mr = fmaxf(mr, __shfl_xor(mr, x, 32));
      float mnew = fmaxf(mrun[r], mr);
      float f  = __expf(mrun[r] - mnew);
      float p0 = __expf(sv[0][r] - mnew);
      float p1 = __expf(sv[1][r] - mnew);
      sv[0][r] = p0; sv[1][r] = p1;
      float rs = p0 + p1;
#pragma unroll
      for (int x = 1; x < 16; x <<= 1) rs += __shfl_xor(rs, x, 32);
      lrun[r] = lrun[r] * f + rs;
      mrun[r] = mnew;
      fac[r] = f;
    }
#pragma unroll
    for (int t = 0; t < 4; ++t)
#pragma unroll
      for (int r = 0; r < 8; ++r) oacc[t][r] *= fac[r];

#pragma unroll
    for (int r = 0; r < 8; ++r) {
      int prow = r + 8 * half;
      lds_p[wave][prow * 32 + l15]      = f2bf(sv[0][r]);
      lds_p[wave][prow * 32 + 16 + l15] = f2bf(sv[1][r]);
    }
    asm volatile("s_wait_dscnt 0" ::: "memory");
    const unsigned short* lp = &lds_p[wave][l15 * 32 + half * 16];
    v16bf ap;
    v8bf* aph = (v8bf*)&ap;
    aph[0] = *(const v8bf*)(lp);
    aph[1] = *(const v8bf*)(lp + 8);

#pragma unroll
    for (int t = 0; t < 4; ++t) {
      const __bf16* vp = vbase + (size_t)(t * 16 + l15) * S_ + j + half * 16;
      v16bf bvv = *(const v16bf*)(vp);
      oacc[t] = wmma_bf16(ap, bvv, oacc[t]);
    }
  }

#pragma unroll
  for (int r = 0; r < 8; ++r) {
    float inv = 1.0f / fmaxf(lrun[r], 1e-30f);
    int ss = qt * 16 + r + 8 * half;
#pragma unroll
    for (int t = 0; t < 4; ++t) {
      int dd = t * 16 + l15;
      O[((size_t)b * S_ + ss) * D_ + h * DH_ + dd] = f2bf(oacc[t][r] * inv);
    }
  }
}

// ----------------------------------------- mask + residual + LayerNorm (fused)
__global__ __launch_bounds__(256) void resid_ln_kernel(
    const float* __restrict__ y, const int* __restrict__ qmask,
    const float* __restrict__ resid, const float* __restrict__ g,
    const float* __restrict__ bta, float* __restrict__ xout,
    unsigned short* __restrict__ xb) {
  const int row = blockIdx.x;
  const int tid = threadIdx.x;
  __shared__ float red[256];
  const float qm = (float)qmask[row];
  const size_t base = (size_t)row * D_;
  float v0 = y[base + tid] * qm + resid[base + tid];
  float v1 = y[base + 256 + tid] * qm + resid[base + 256 + tid];
  red[tid] = v0 + v1;
  __syncthreads();
  for (int s = 128; s > 0; s >>= 1) {
    if (tid < s) red[tid] += red[tid + s];
    __syncthreads();
  }
  const float mean = red[0] * (1.0f / D_);
  __syncthreads();
  float d0 = v0 - mean, d1 = v1 - mean;
  red[tid] = d0 * d0 + d1 * d1;
  __syncthreads();
  for (int s = 128; s > 0; s >>= 1) {
    if (tid < s) red[tid] += red[tid + s];
    __syncthreads();
  }
  const float rstd = rsqrtf(red[0] * (1.0f / D_) + 1e-5f);
  float o0 = d0 * rstd * g[tid] + bta[tid];
  float o1 = d1 * rstd * g[tid + 256] + bta[tid + 256];
  xout[base + tid] = o0;
  xout[base + 256 + tid] = o1;
  xb[base + tid] = f2bf(o0);
  xb[base + 256 + tid] = f2bf(o1);
}

// ============================================================================
extern "C" void kernel_launch(void* const* d_in, const int* in_sizes, int n_in,
                              void* d_out, int out_size, void* d_ws,
                              size_t ws_size, hipStream_t stream) {
  (void)in_sizes; (void)n_in; (void)out_size; (void)ws_size;
  const float* query = (const float*)d_in[0];
  const float* key   = (const float*)d_in[1];
  const int*   qmask = (const int*)d_in[2];
  const int*   kmask = (const int*)d_in[3];
  const float* Wq = (const float*)d_in[4];  const float* bq = (const float*)d_in[5];
  const float* Wk = (const float*)d_in[6];  const float* bk = (const float*)d_in[7];
  const float* Wv = (const float*)d_in[8];  const float* bv = (const float*)d_in[9];
  const float* Wo = (const float*)d_in[10]; const float* bo = (const float*)d_in[11];
  const float* ln1g = (const float*)d_in[12]; const float* ln1b = (const float*)d_in[13];
  const float* W1 = (const float*)d_in[14]; const float* b1 = (const float*)d_in[15];
  const float* W2 = (const float*)d_in[16]; const float* b2 = (const float*)d_in[17];
  const float* ln2g = (const float*)d_in[18]; const float* ln2b = (const float*)d_in[19];

  const size_t BSD = (size_t)BS_ * D_;
  const size_t BSF = (size_t)BS_ * FF_;
  const size_t WDD = (size_t)L_ * D_ * D_;
  const size_t WFD = (size_t)L_ * FF_ * D_;

  char* ws = (char*)d_ws;
  size_t off = 0;
  auto take = [&](size_t bytes) -> char* {
    char* p = ws + off;
    off += (bytes + 255) & ~(size_t)255;
    return p;
  };
  unsigned short* xbf  = (unsigned short*)take(BSD * 2);
  unsigned short* kybf = (unsigned short*)take(BSD * 2);
  unsigned short* qbf  = (unsigned short*)take(BSD * 2);
  unsigned short* kcbf = (unsigned short*)take(BSD * 2);
  unsigned short* vtbf = (unsigned short*)take(BSD * 2);
  unsigned short* obf  = (unsigned short*)take(BSD * 2);
  unsigned short* hbf  = (unsigned short*)take(BSF * 2);
  unsigned short* wqb  = (unsigned short*)take(WDD * 2);
  unsigned short* wkb  = (unsigned short*)take(WDD * 2);
  unsigned short* wvb  = (unsigned short*)take(WDD * 2);
  unsigned short* wob  = (unsigned short*)take(WDD * 2);
  unsigned short* w1b  = (unsigned short*)take(WFD * 2);
  unsigned short* w2b  = (unsigned short*)take(WFD * 2);
  float* gemmf = (float*)take(BSF * 4);
  float* xcur  = (float*)take(BSD * 4);

  auto cvt = [&](const float* src, unsigned short* dst, size_t n) {
    cvt_bf16_kernel<<<(unsigned)((n + 255) / 256), 256, 0, stream>>>(src, dst,
                                                                     (long long)n);
  };
  cvt(query, xbf, BSD);
  cvt(key, kybf, BSD);
  cvt(Wq, wqb, WDD); cvt(Wk, wkb, WDD); cvt(Wv, wvb, WDD); cvt(Wo, wob, WDD);
  cvt(W1, w1b, WFD); cvt(W2, w2b, WFD);
  (void)hipMemcpyAsync(xcur, query, BSD * 4, hipMemcpyDeviceToDevice, stream);

  const float qscale = 0.125f;  // DH^-0.5 = 1/8
  const dim3 gD(D_ / 256, BS_ / 32);   // N=512 GEMMs: (2,128)
  const dim3 gF(FF_ / 256, BS_ / 32);  // N=2048 GEMM: (8,128)

  for (int i = 0; i < L_; ++i) {
    const __bf16* wq = (const __bf16*)(wqb + (size_t)i * D_ * D_);
    const __bf16* wk = (const __bf16*)(wkb + (size_t)i * D_ * D_);
    const __bf16* wv = (const __bf16*)(wvb + (size_t)i * D_ * D_);
    const __bf16* wo = (const __bf16*)(wob + (size_t)i * D_ * D_);
    const __bf16* w1 = (const __bf16*)(w1b + (size_t)i * FF_ * D_);
    const __bf16* w2 = (const __bf16*)(w2b + (size_t)i * D_ * FF_);

    gemm_wmma<1><<<gD, 256, 0, stream>>>((const __bf16*)xbf,  wq, bq + i * D_,
                                         nullptr, qbf,  BS_, D_, D_, qscale);
    gemm_wmma<2><<<gD, 256, 0, stream>>>((const __bf16*)kybf, wk, bk + i * D_,
                                         nullptr, kcbf, BS_, D_, D_, 1.f);
    gemm_wmma<3><<<gD, 256, 0, stream>>>((const __bf16*)kybf, wv, bv + i * D_,
                                         nullptr, vtbf, BS_, D_, D_, 1.f);
    attn_wmma<<<(B_ * H_ * (S_ / 16)) / 8, 256, 0, stream>>>(
        (const __bf16*)qbf, (const __bf16*)kcbf, (const __bf16*)vtbf, kmask, obf);
    gemm_wmma<4><<<gD, 256, 0, stream>>>((const __bf16*)obf, wo, bo + i * D_,
                                         gemmf, nullptr, BS_, D_, D_, 1.f);
    resid_ln_kernel<<<BS_, 256, 0, stream>>>(gemmf, qmask, xcur, ln1g + i * D_,
                                             ln1b + i * D_, xcur, xbf);
    gemm_wmma<5><<<gF, 256, 0, stream>>>((const __bf16*)xbf, w1, b1 + i * FF_,
                                         nullptr, hbf, BS_, FF_, D_, 1.f);
    gemm_wmma<4><<<gD, 256, 0, stream>>>((const __bf16*)hbf, w2, b2 + i * D_,
                                         gemmf, nullptr, BS_, D_, FF_, 1.f);
    resid_ln_kernel<<<BS_, 256, 0, stream>>>(gemmf, qmask, xcur, ln2g + i * D_,
                                             ln2b + i * D_, xcur, xbf);
  }
  (void)hipMemcpyAsync(d_out, xcur, BSD * 4, hipMemcpyDeviceToDevice, stream);
}
